// CompressiveDecoder_6433861009888
// MI455X (gfx1250) — matrix operand order repaired and checked by reading, not compiled, not run on hardware
//
#include <hip/hip_runtime.h>
#include <math.h>

// ---------------------------------------------------------------------------
// Problem constants (match reference)
// ---------------------------------------------------------------------------
#define Bc   8
#define Sc   512
#define Lc   256
#define Dc   512
#define Hc   8
#define DHc  64
#define Nc   4
#define Vc   512
#define FFc  2048
#define TRK  4
#define BSr  (Bc * Sc)   // 4096 rows of x
#define BLr  (Bc * Lc)   // 2048 rows of latent

// ---------------------------------------------------------------------------
// WMMA vector types (gfx1250, wave32)
// ---------------------------------------------------------------------------
typedef __attribute__((ext_vector_type(16))) _Float16 v16h;
typedef __attribute__((ext_vector_type(8)))  _Float16 v8h;
typedef __attribute__((ext_vector_type(8)))  float    v8f;
typedef __attribute__((ext_vector_type(4)))  int      v4i;

#if __has_builtin(__builtin_amdgcn_global_load_async_to_lds_b128) && \
    __has_builtin(__builtin_amdgcn_s_wait_asynccnt)
#define USE_ASYNC_LDS 1
#else
#define USE_ASYNC_LDS 0
#endif

__device__ __forceinline__ v16h join8(v8h lo, v8h hi) {
  v16h r;
#pragma unroll
  for (int i = 0; i < 8; ++i) { r[i] = lo[i]; r[i + 8] = hi[i]; }
  return r;
}

__device__ __forceinline__ v8f wmma_f16(v16h a, v16h b, v8f c) {
  return __builtin_amdgcn_wmma_f32_16x16x32_f16(
      /*neg_a=*/false, a, /*neg_b=*/false, b,
      /*c_mod=*/(short)0, c, /*reuse_a=*/false, /*reuse_b=*/false);
}

// Move 16 consecutive f16 (32B) global -> LDS, async on gfx1250 if available.
__device__ __forceinline__ void stage16(const _Float16* g, _Float16* l) {
#if USE_ASYNC_LDS
  __builtin_amdgcn_global_load_async_to_lds_b128(
      (__attribute__((address_space(1))) v4i*)g,
      (__attribute__((address_space(3))) v4i*)l, 0, 0);
  __builtin_amdgcn_global_load_async_to_lds_b128(
      (__attribute__((address_space(1))) v4i*)(g + 8),
      (__attribute__((address_space(3))) v4i*)(l + 8), 0, 0);
#else
  *(v8h*)l       = *(const v8h*)g;
  *(v8h*)(l + 8) = *(const v8h*)(g + 8);
#endif
}

__device__ __forceinline__ float gelu_f(float x) {
  return 0.5f * x * (1.0f + erff(x * 0.70710678118654752f));
}

// ---------------------------------------------------------------------------
// Elementwise fp32 -> f16
// ---------------------------------------------------------------------------
__global__ __launch_bounds__(256) void cvt_h_k(const float* __restrict__ x,
                                               _Float16* __restrict__ y) {
  int i = blockIdx.x * 256 + threadIdx.x;
  y[i] = (_Float16)x[i];
}

// Transpose + convert: Wt[n*K + k] = (f16) W[k*N + n]
__global__ __launch_bounds__(256) void cvt_t_h_k(const float* __restrict__ W,
                                                 _Float16* __restrict__ Wt,
                                                 int K, int Nn) {
  int i = blockIdx.x * 256 + threadIdx.x;   // over K*Nn
  int n = i / K;
  int k = i - n * K;
  Wt[i] = (_Float16)W[(size_t)k * Nn + n];
}

// ---------------------------------------------------------------------------
// Embedding + sinusoidal positional encoding (fp32 residual stream)
// ---------------------------------------------------------------------------
__global__ __launch_bounds__(256) void embed_pe_k(
    const int* __restrict__ trg, const float* __restrict__ emb,
    float* __restrict__ x) {
  int idx = blockIdx.x * 256 + threadIdx.x;       // over BSr*Dc
  int d   = idx & (Dc - 1);
  int rs  = idx >> 9;                             // Dc == 512
  int s   = rs & (Sc - 1);
  int b   = rs >> 9;                              // Sc == 512
  int tok = trg[b * (TRK * Sc) + s];
  float k2  = (float)(d & ~1);
  float ang = (float)s * __expf(-(9.210340371976184f / (float)Dc) * k2);
  float pe  = (d & 1) ? __cosf(ang) : __sinf(ang);
  x[idx] = emb[(size_t)tok * Dc + d] + pe;
}

// ---------------------------------------------------------------------------
// LayerNorm: one row (Dc=512) per 256-thread block; f16 output (feeds GEMM A)
// ---------------------------------------------------------------------------
__global__ __launch_bounds__(256) void layernorm_k(
    const float* __restrict__ x, const float* __restrict__ g,
    const float* __restrict__ bt, _Float16* __restrict__ y) {
  __shared__ float redS[8], redQ[8];
  int row = blockIdx.x;
  const float* xr = x + (size_t)row * Dc;
  float s = 0.f, ss = 0.f;
  for (int d = threadIdx.x; d < Dc; d += 256) {
    float v = xr[d]; s += v; ss += v * v;
  }
#pragma unroll
  for (int off = 16; off; off >>= 1) {
    s  += __shfl_xor(s,  off, 32);
    ss += __shfl_xor(ss, off, 32);
  }
  if ((threadIdx.x & 31) == 0) {
    redS[threadIdx.x >> 5] = s; redQ[threadIdx.x >> 5] = ss;
  }
  __syncthreads();
  s = 0.f; ss = 0.f;
#pragma unroll
  for (int i = 0; i < 8; ++i) { s += redS[i]; ss += redQ[i]; }
  float mean = s * (1.0f / Dc);
  float var  = ss * (1.0f / Dc) - mean * mean;
  float rstd = rsqrtf(var + 1e-5f);
  _Float16* yr = y + (size_t)row * Dc;
  for (int d = threadIdx.x; d < Dc; d += 256)
    yr[d] = (_Float16)((xr[d] - mean) * rstd * g[d] + bt[d]);
}

// ---------------------------------------------------------------------------
// Tiled GEMM:  C[M,N] = act(A[M,K] * Wt[N,K]^T + bias) (+ resid)
// A, Wt are f16 in memory; tiles async-staged to LDS; wmma f32.16x16x32.f16.
// Block 256 threads = 8 waves; BM=BN=64, BK=64; wave -> 16x32 of C.
// All M,N,K here are multiples of 64 (no edge handling).
// ---------------------------------------------------------------------------
#define BM 64
#define BN 64
#define BK 64
#define STH 72   // padded halves per LDS tile row (144B, 16B-aligned chunks)

__global__ __launch_bounds__(256) void gemm_f16wmma_k(
    const _Float16* __restrict__ A, const _Float16* __restrict__ Wt,
    const float* __restrict__ bias, const float* __restrict__ resid,
    void* __restrict__ Cout, int M, int Nn, int K, int act, int outHalf) {
  __shared__ _Float16 As[BM * STH];
  __shared__ _Float16 Bs[BN * STH];

  int tid  = threadIdx.x;
  int lane = tid & 31;
  int wave = tid >> 5;        // 0..7
  int wm   = wave & 3;        // 4 row tiles of 16
  int wn   = wave >> 2;       // 2 col strips of 32
  int bm0  = blockIdx.x * BM;
  int bn0  = blockIdx.y * BN;
  int l15  = lane & 15;
  int g    = lane >> 4;       // 16-lane half select
  int sr   = tid >> 2;        // staging row 0..63
  int sc   = (tid & 3) * 16;  // staging col 0/16/32/48

  v8f acc0 = {};
  v8f acc1 = {};

  for (int k0 = 0; k0 < K; k0 += BK) {
    stage16(&A [(size_t)(bm0 + sr) * K + k0 + sc], &As[sr * STH + sc]);
    stage16(&Wt[(size_t)(bn0 + sr) * K + k0 + sc], &Bs[sr * STH + sc]);
#if USE_ASYNC_LDS
    __builtin_amdgcn_s_wait_asynccnt(0);
#endif
    __syncthreads();

#pragma unroll
    for (int s = 0; s < 2; ++s) {   // two K=32 sub-steps of the 64-wide stage
      const _Float16* ap = &As[(wm * 16 + l15) * STH + 32 * s + g * 16];
      v16h afrag = join8(*(const v8h*)ap, *(const v8h*)(ap + 8));
      {
        const _Float16* bp = &Bs[(wn * 32 + l15) * STH + 32 * s + g * 16];
        v16h bfrag = join8(*(const v8h*)bp, *(const v8h*)(bp + 8));
        acc0 = wmma_f16(afrag, bfrag, acc0);
      }
      {
        const _Float16* bp = &Bs[(wn * 32 + 16 + l15) * STH + 32 * s + g * 16];
        v16h bfrag = join8(*(const v8h*)bp, *(const v8h*)(bp + 8));
        acc1 = wmma_f16(afrag, bfrag, acc1);
      }
    }
    __syncthreads();
  }

  // Epilogue: C layout — lane half g holds rows r+8g, col = lane&15
  int m0 = bm0 + wm * 16 + g * 8;
  int n0 = bn0 + wn * 32 + l15;
#pragma unroll
  for (int r = 0; r < 8; ++r) {
    size_t base = (size_t)(m0 + r) * Nn;
    float v0 = acc0[r], v1 = acc1[r];
    if (bias) { v0 += bias[n0]; v1 += bias[n0 + 16]; }
    if (act == 1) { v0 = gelu_f(v0); v1 = gelu_f(v1); }
    if (resid) { v0 += resid[base + n0]; v1 += resid[base + n0 + 16]; }
    if (outHalf) {
      ((_Float16*)Cout)[base + n0]      = (_Float16)v0;
      ((_Float16*)Cout)[base + n0 + 16] = (_Float16)v1;
    } else {
      ((float*)Cout)[base + n0]      = v0;
      ((float*)Cout)[base + n0 + 16] = v1;
    }
  }
}

// ---------------------------------------------------------------------------
// Flash attention (WMMA, f16 in / f16 out): one wave = 16 query rows, (b,h).
// Scores via 4x wmma per 32-key step; online softmax with 16-lane shuffles;
// P converted C-layout -> A-layout through per-wave LDS; P*V via 4x wmma.
// ---------------------------------------------------------------------------
#define STP 40

__global__ __launch_bounds__(128) void flash_attn_k(
    const _Float16* __restrict__ Q, const _Float16* __restrict__ Kb,
    const _Float16* __restrict__ Vb, _Float16* __restrict__ O,
    int Sq, int Skv, int causal) {
  __shared__ _Float16 Pls[4][16 * STP];   // per-wave 16x32 P tile (padded)

  int lane = threadIdx.x & 31;
  int wave = threadIdx.x >> 5;
  int l15  = lane & 15;
  int g    = lane >> 4;
  int m0   = blockIdx.x * 64 + wave * 16;
  int h    = blockIdx.y;
  int b    = blockIdx.z;
  const float scale = 0.125f;   // DH^-0.5

  // Q fragments: A layout, row = l15; aligned v8h chunks.
  const _Float16* qrow = Q + ((size_t)(b * Sq + m0 + l15)) * Dc + h * DHc;
  v16h aQ[2];
#pragma unroll
  for (int f = 0; f < 2; ++f)
    aQ[f] = join8(*(const v8h*)(qrow + 16 * g + 32 * f),
                  *(const v8h*)(qrow + 8 + 16 * g + 32 * f));

  v8f m_i, l_i, o0 = {}, o1 = {}, o2 = {}, o3 = {};
#pragma unroll
  for (int r = 0; r < 8; ++r) { m_i[r] = -1e30f; l_i[r] = 0.f; }

  int jend = causal ? (m0 + 16) : Skv;
  for (int j0 = 0; j0 < jend; j0 += 32) {
    // ---- scores for keys [j0, j0+32) as two 16-key tiles
    v8f s0 = {}, s1 = {};
#pragma unroll
    for (int f = 0; f < 2; ++f) {
      const _Float16* kr0 =
          Kb + ((size_t)(b * Skv + j0 + l15)) * Dc + h * DHc + 16 * g + 32 * f;
      const _Float16* kr1 =
          Kb + ((size_t)(b * Skv + j0 + 16 + l15)) * Dc + h * DHc + 16 * g + 32 * f;
      v16h bf0 = join8(*(const v8h*)kr0, *(const v8h*)(kr0 + 8));
      v16h bf1 = join8(*(const v8h*)kr1, *(const v8h*)(kr1 + 8));
      s0 = wmma_f16(aQ[f], bf0, s0);
      s1 = wmma_f16(aQ[f], bf1, s1);
    }
#pragma unroll
    for (int r = 0; r < 8; ++r) { s0[r] *= scale; s1[r] *= scale; }

    if (causal) {
#pragma unroll
      for (int r = 0; r < 8; ++r) {
        int i = m0 + r + 8 * g;
        if (j0 + l15 > i)      s0[r] = -1e9f;
        if (j0 + 16 + l15 > i) s1[r] = -1e9f;
      }
    }

    // ---- online softmax: row reductions stay within each 16-lane half
    v8f tmax;
#pragma unroll
    for (int r = 0; r < 8; ++r) tmax[r] = fmaxf(s0[r], s1[r]);
#pragma unroll
    for (int off = 1; off < 16; off <<= 1)
#pragma unroll
      for (int r = 0; r < 8; ++r)
        tmax[r] = fmaxf(tmax[r], __shfl_xor(tmax[r], off, 32));

    v8f mnew, alpha, p0, p1, rs;
#pragma unroll
    for (int r = 0; r < 8; ++r) {
      mnew[r]  = fmaxf(m_i[r], tmax[r]);
      alpha[r] = __expf(m_i[r] - mnew[r]);
      p0[r]    = __expf(s0[r] - mnew[r]);
      p1[r]    = __expf(s1[r] - mnew[r]);
      rs[r]    = p0[r] + p1[r];
    }
#pragma unroll
    for (int off = 1; off < 16; off <<= 1)
#pragma unroll
      for (int r = 0; r < 8; ++r) rs[r] += __shfl_xor(rs[r], off, 32);
#pragma unroll
    for (int r = 0; r < 8; ++r) {
      l_i[r] = l_i[r] * alpha[r] + rs[r];
      m_i[r] = mnew[r];
      o0[r] *= alpha[r]; o1[r] *= alpha[r];
      o2[r] *= alpha[r]; o3[r] *= alpha[r];
    }

    // ---- P: C layout -> LDS -> A layout (same-wave DS ops stay in order)
    _Float16* pl = &Pls[wave][0];
#pragma unroll
    for (int r = 0; r < 8; ++r) {
      pl[(r + 8 * g) * STP + l15]      = (_Float16)p0[r];
      pl[(r + 8 * g) * STP + 16 + l15] = (_Float16)p1[r];
    }
    const _Float16* pr = &Pls[wave][l15 * STP];
    v16h aP = join8(*(const v8h*)(pr + 16 * g), *(const v8h*)(pr + 8 + 16 * g));

    // ---- accumulate P * V (4 DH chunks of 16)
#pragma unroll
    for (int d = 0; d < 4; ++d) {
      v16h bV;
      const _Float16* vcol =
          Vb + ((size_t)(b * Skv + j0 + 16 * g)) * Dc + h * DHc + d * 16 + l15;
#pragma unroll
      for (int e = 0; e < 16; ++e) bV[e] = vcol[(size_t)e * Dc];
      if (d == 0) o0 = wmma_f16(aP, bV, o0);
      else if (d == 1) o1 = wmma_f16(aP, bV, o1);
      else if (d == 2) o2 = wmma_f16(aP, bV, o2);
      else o3 = wmma_f16(aP, bV, o3);
    }
  }

  // ---- normalize and store (f16, feeds the output-projection GEMM)
#pragma unroll
  for (int r = 0; r < 8; ++r) {
    float inv = 1.0f / l_i[r];
    int i = m0 + r + 8 * g;
    _Float16* orow = O + ((size_t)(b * Sq + i)) * Dc + h * DHc;
    orow[0 * 16 + l15] = (_Float16)(o0[r] * inv);
    orow[1 * 16 + l15] = (_Float16)(o1[r] * inv);
    orow[2 * 16 + l15] = (_Float16)(o2[r] * inv);
    orow[3 * 16 + l15] = (_Float16)(o3[r] * inv);
  }
}

// ---------------------------------------------------------------------------
// log_softmax over V=512 per row; writes transposed [B,S,V,TRK] at track t.
// ---------------------------------------------------------------------------
__global__ __launch_bounds__(256) void logsoftmax_k(
    const float* __restrict__ X, float* __restrict__ out, int t) {
  __shared__ float redM[8], redS[8];
  int row = blockIdx.x;
  const float* xr = X + (size_t)row * Vc;
  float mx = -1e30f;
  for (int v = threadIdx.x; v < Vc; v += 256) mx = fmaxf(mx, xr[v]);
#pragma unroll
  for (int off = 16; off; off >>= 1) mx = fmaxf(mx, __shfl_xor(mx, off, 32));
  if ((threadIdx.x & 31) == 0) redM[threadIdx.x >> 5] = mx;
  __syncthreads();
  mx = redM[0];
#pragma unroll
  for (int i = 1; i < 8; ++i) mx = fmaxf(mx, redM[i]);
  float s = 0.f;
  for (int v = threadIdx.x; v < Vc; v += 256) s += __expf(xr[v] - mx);
#pragma unroll
  for (int off = 16; off; off >>= 1) s += __shfl_xor(s, off, 32);
  if ((threadIdx.x & 31) == 0) redS[threadIdx.x >> 5] = s;
  __syncthreads();
  s = 0.f;
#pragma unroll
  for (int i = 0; i < 8; ++i) s += redS[i];
  float lse = mx + logf(s);
  for (int v = threadIdx.x; v < Vc; v += 256)
    out[((size_t)row * Vc + v) * TRK + t] = xr[v] - lse;
}

// ---------------------------------------------------------------------------
// Host orchestration
// ---------------------------------------------------------------------------
extern "C" void kernel_launch(void* const* d_in, const int* in_sizes, int n_in,
                              void* d_out, int out_size, void* d_ws,
                              size_t ws_size, hipStream_t stream) {
  (void)in_sizes; (void)n_in; (void)out_size; (void)ws_size;

  const int*   trg     = (const int*)d_in[0];
  const float* latent  = (const float*)d_in[1];
  const float* embed   = (const float*)d_in[6];
  const float* ln1g    = (const float*)d_in[7];
  const float* ln1b    = (const float*)d_in[8];
  const float* selfW   = (const float*)d_in[9];
  const float* selfB   = (const float*)d_in[10];
  const float* ln2g    = (const float*)d_in[11];
  const float* ln2b    = (const float*)d_in[12];
  const float* crossW  = (const float*)d_in[13];
  const float* crossB  = (const float*)d_in[14];
  const float* ln3g    = (const float*)d_in[15];
  const float* ln3b    = (const float*)d_in[16];
  const float* w1      = (const float*)d_in[17];
  const float* b1      = (const float*)d_in[18];
  const float* w2      = (const float*)d_in[19];
  const float* b2      = (const float*)d_in[20];
  const float* genW    = (const float*)d_in[21];
  const float* genb    = (const float*)d_in[22];
  float* out = (float*)d_out;

  // Workspace carve-up (mixed fp32 / f16), 256B aligned chunks
  char* p = (char*)d_ws;
  auto carve = [&](size_t bytes) {
    void* r = (void*)p;
    p += (bytes + 255) & ~(size_t)255;
    return r;
  };
  float*     xA     = (float*)    carve((size_t)BSr * Dc * 4);
  float*     xB     = (float*)    carve((size_t)BSr * Dc * 4);
  _Float16*  xnh    = (_Float16*) carve((size_t)BSr * Dc * 2);
  _Float16*  qh     = (_Float16*) carve((size_t)BSr * Dc * 2);
  _Float16*  kh     = (_Float16*) carve((size_t)BSr * Dc * 2);
  _Float16*  vh     = (_Float16*) carve((size_t)BSr * Dc * 2);
  _Float16*  aoh    = (_Float16*) carve((size_t)BSr * Dc * 2);
  _Float16*  mhh    = (_Float16*) carve((size_t)BSr * FFc * 2);
  _Float16*  lath   = (_Float16*) carve((size_t)BLr * Dc * 2);
  _Float16*  wth    = (_Float16*) carve((size_t)Dc * FFc * 2);
  float*     logits = (float*)    carve((size_t)BSr * Vc * 4);

  // One GEMM = transpose+convert weights to f16 [N,K], then WMMA GEMM.
  auto gemmW = [&](const _Float16* Ah, const float* W, const float* bias,
                   const float* resid, void* C, int M, int Nn, int K,
                   int act, int outHalf) {
    cvt_t_h_k<<<((size_t)K * Nn) / 256, 256, 0, stream>>>(W, wth, K, Nn);
    dim3 grid(M / BM, Nn / BN);
    gemm_f16wmma_k<<<grid, 256, 0, stream>>>(Ah, wth, bias, resid, C,
                                             M, Nn, K, act, outHalf);
  };

  for (int t = 0; t < TRK; ++t) {
    const float* lat_t = latent + (size_t)t * Bc * Lc * Dc;
    const float* emb_t = embed + (size_t)t * Vc * Dc;

    cvt_h_k<<<((size_t)BLr * Dc) / 256, 256, 0, stream>>>(lat_t, lath);
    embed_pe_k<<<((size_t)BSr * Dc) / 256, 256, 0, stream>>>(
        trg + (size_t)t * Sc, emb_t, xA);

    float* xcur = xA;
    float* xalt = xB;

    for (int i = 0; i < Nc; ++i) {
      size_t li = (size_t)t * Nc + i;
      const float* sW = selfW + li * 4 * (size_t)Dc * Dc;
      const float* sb = selfB + li * 4 * (size_t)Dc;
      const float* cW = crossW + li * 4 * (size_t)Dc * Dc;
      const float* cb = crossB + li * 4 * (size_t)Dc;

      // --- self attention (pre-norm, causal) ---
      layernorm_k<<<BSr, 256, 0, stream>>>(xcur, ln1g + li * Dc, ln1b + li * Dc, xnh);
      gemmW(xnh, sW + 0 * (size_t)Dc * Dc, sb + 0 * Dc, nullptr, qh, BSr, Dc, Dc, 0, 1);
      gemmW(xnh, sW + 1 * (size_t)Dc * Dc, sb + 1 * Dc, nullptr, kh, BSr, Dc, Dc, 0, 1);
      gemmW(xnh, sW + 2 * (size_t)Dc * Dc, sb + 2 * Dc, nullptr, vh, BSr, Dc, Dc, 0, 1);
      flash_attn_k<<<dim3(Sc / 64, Hc, Bc), 128, 0, stream>>>(
          qh, kh, vh, aoh, Sc, Sc, /*causal=*/1);
      gemmW(aoh, sW + 3 * (size_t)Dc * Dc, sb + 3 * Dc, xcur, xalt, BSr, Dc, Dc, 0, 0);
      { float* tmp = xcur; xcur = xalt; xalt = tmp; }

      // --- cross attention to latent (pre-norm q, raw kv) ---
      layernorm_k<<<BSr, 256, 0, stream>>>(xcur, ln2g + li * Dc, ln2b + li * Dc, xnh);
      gemmW(xnh,  cW + 0 * (size_t)Dc * Dc, cb + 0 * Dc, nullptr, qh, BSr, Dc, Dc, 0, 1);
      gemmW(lath, cW + 1 * (size_t)Dc * Dc, cb + 1 * Dc, nullptr, kh, BLr, Dc, Dc, 0, 1);
      gemmW(lath, cW + 2 * (size_t)Dc * Dc, cb + 2 * Dc, nullptr, vh, BLr, Dc, Dc, 0, 1);
      flash_attn_k<<<dim3(Sc / 64, Hc, Bc), 128, 0, stream>>>(
          qh, kh, vh, aoh, Sc, Lc, /*causal=*/0);
      gemmW(aoh, cW + 3 * (size_t)Dc * Dc, cb + 3 * Dc, xcur, xalt, BSr, Dc, Dc, 0, 0);
      { float* tmp = xcur; xcur = xalt; xalt = tmp; }

      // --- GELU MLP (pre-norm) ---
      layernorm_k<<<BSr, 256, 0, stream>>>(xcur, ln3g + li * Dc, ln3b + li * Dc, xnh);
      gemmW(xnh, w1 + li * (size_t)Dc * FFc, b1 + li * FFc, nullptr, mhh,
            BSr, FFc, Dc, /*gelu=*/1, 1);
      gemmW(mhh, w2 + li * (size_t)FFc * Dc, b2 + li * Dc, xcur, xalt,
            BSr, Dc, FFc, 0, 0);
      { float* tmp = xcur; xcur = xalt; xalt = tmp; }
    }

    // --- generator + log_softmax into transposed output slot t ---
    cvt_h_k<<<((size_t)BSr * Dc) / 256, 256, 0, stream>>>(xcur, xnh);
    gemmW(xnh, genW + (size_t)t * Dc * Vc, genb + (size_t)t * Vc, nullptr,
          logits, BSr, Vc, Dc, 0, 0);
    logsoftmax_k<<<BSr, 256, 0, stream>>>(logits, out, t);
  }
}
